// Model_43997644980794
// MI455X (gfx1250) — matrix operand order repaired and checked by reading, not compiled, not run on hardware
//
#include <hip/hip_runtime.h>
#include <stdint.h>
#include <math.h>

#define B_SZ      16
#define SEQ_LEN   512
#define LABEL_LEN 256
#define PRED_LEN  256
#define DEC_LEN   512
#define ENC_IN    21
#define C_OUT     21
#define MARK      4
#define D_MODEL   512
#define N_HEADS   8
#define D_FF      2048
#define HEAD_DIM  64

typedef __attribute__((ext_vector_type(16))) _Float16 v16h;
typedef __attribute__((ext_vector_type(8)))  _Float16 v8h;
typedef __attribute__((ext_vector_type(2)))  _Float16 v2h;
typedef __attribute__((ext_vector_type(8)))  float    v8f;

// ---------------------------------------------------------------- utilities
__global__ void k_f32_to_f16(const float* __restrict__ s, _Float16* __restrict__ d, int n) {
  int i = blockIdx.x * blockDim.x + threadIdx.x;
  if (i < n) d[i] = (_Float16)s[i];
}

// zero-padded f16 copy of [K,Nsrc] into [K,Npad] (final projection operand)
__global__ void k_padw(const float* __restrict__ src, _Float16* __restrict__ dst,
                       int K, int Nsrc, int Npad) {
  int idx = blockIdx.x * blockDim.x + threadIdx.x;
  if (idx >= K * Npad) return;
  int k = idx / Npad, n = idx % Npad;
  dst[idx] = (n < Nsrc) ? (_Float16)src[k * Nsrc + n] : (_Float16)0.f;
}

// per-(batch,channel) mean / std over time (one wave each)
__global__ __launch_bounds__(32) void k_stats(const float* __restrict__ x,
                                              float* __restrict__ mean,
                                              float* __restrict__ stdv) {
  int bi = blockIdx.x;
  int b = bi / ENC_IN, i = bi % ENC_IN;
  float s = 0.f, ss = 0.f;
  for (int t = threadIdx.x; t < SEQ_LEN; t += 32) {
    float v = x[((size_t)b * SEQ_LEN + t) * ENC_IN + i];
    s += v; ss += v * v;
  }
#pragma unroll
  for (int m = 16; m >= 1; m >>= 1) { s += __shfl_xor(s, m); ss += __shfl_xor(ss, m); }
  if (threadIdx.x == 0) {
    float mu = s * (1.f / SEQ_LEN);
    mean[bi] = mu;
    stdv[bi] = sqrtf(ss * (1.f / SEQ_LEN) - mu * mu + 1e-5f);
  }
}

// projector circular conv over the 21 "positions": s[b,i] (one wave each)
__global__ __launch_bounds__(32) void k_projconv(const float* __restrict__ x,
                                                 const float* __restrict__ conv,  // [1,512,3]
                                                 float* __restrict__ s) {
  int bi = blockIdx.x;
  int b = bi / ENC_IN, i = bi % ENC_IN;
  int im1 = (i + ENC_IN - 1) % ENC_IN, ip1 = (i + 1) % ENC_IN;
  float acc = 0.f;
  for (int t = threadIdx.x; t < SEQ_LEN; t += 32) {
    const float* xr = x + ((size_t)b * SEQ_LEN + t) * ENC_IN;
    const float* cw = conv + t * 3;
    acc += xr[im1] * cw[0] + xr[i] * cw[1] + xr[ip1] * cw[2];
  }
#pragma unroll
  for (int m = 16; m >= 1; m >>= 1) acc += __shfl_xor(acc, m);
  if (threadIdx.x == 0) s[bi] = acc;
}

// tiny MLP 42 -> 128 -> 128 -> outDim  (one block per batch)
__global__ __launch_bounds__(128) void k_projmlp(const float* __restrict__ s,
                                                 const float* __restrict__ stat,
                                                 const float* __restrict__ w0, const float* __restrict__ b0,
                                                 const float* __restrict__ w1, const float* __restrict__ b1,
                                                 const float* __restrict__ w2,
                                                 float* __restrict__ out, int outDim, int doExp) {
  __shared__ float h[2 * ENC_IN], h1[128], h2[128];
  int b = blockIdx.x, t = threadIdx.x;
  if (t < 2 * ENC_IN) h[t] = (t < ENC_IN) ? s[b * ENC_IN + t] : stat[b * ENC_IN + (t - ENC_IN)];
  __syncthreads();
  {
    float a = b0[t];
    for (int j = 0; j < 2 * ENC_IN; ++j) a += h[j] * w0[j * 128 + t];
    h1[t] = fmaxf(a, 0.f);
  }
  __syncthreads();
  {
    float a = b1[t];
    for (int j = 0; j < 128; ++j) a += h1[j] * w1[j * 128 + t];
    h2[t] = fmaxf(a, 0.f);
  }
  __syncthreads();
  for (int o = t; o < outDim; o += 128) {
    float a = 0.f;
    for (int j = 0; j < 128; ++j) a += h2[j] * w2[j * outDim + o];
    out[b * outDim + o] = doExp ? __expf(a) : a;
  }
}

// embedding: circular conv(x_n) + mark@temp + POS   (decMode builds x_dec_new on the fly)
__global__ void k_embed(const float* __restrict__ x_enc, const float* __restrict__ mean,
                        const float* __restrict__ stdv, const float* __restrict__ conv,  // [512,21,3]
                        const float* __restrict__ temp,                                  // [4,512]
                        const float* __restrict__ mark,
                        float* __restrict__ outF, _Float16* __restrict__ outH, int decMode) {
  int b = blockIdx.y;
  int idx = blockIdx.x * blockDim.x + threadIdx.x;
  if (idx >= SEQ_LEN * D_MODEL) return;
  int l = idx / D_MODEL, d = idx % D_MODEL;
  float acc = 0.f;
  const float* mu = mean + b * ENC_IN;
  const float* sd = stdv + b * ENC_IN;
#pragma unroll
  for (int dw = 0; dw < 3; ++dw) {
    int t = l + dw - 1;
    t = (t + SEQ_LEN) % SEQ_LEN;
    if (decMode && t >= LABEL_LEN) continue;  // zero rows of x_dec_new
    int trow = decMode ? (SEQ_LEN - LABEL_LEN + t) : t;
    const float* xr = x_enc + ((size_t)b * SEQ_LEN + trow) * ENC_IN;
    const float* cw = conv + (size_t)d * ENC_IN * 3 + dw;
    for (int i = 0; i < ENC_IN; ++i)
      acc += ((xr[i] - mu[i]) / sd[i]) * cw[i * 3];
  }
#pragma unroll
  for (int m = 0; m < MARK; ++m)
    acc += mark[((size_t)b * SEQ_LEN + l) * MARK + m] * temp[m * D_MODEL + d];
  int e = d & ~1;
  float div = __expf(-(float)e * (logf(10000.0f) / (float)D_MODEL));
  float ang = (float)l * div;
  acc += (d & 1) ? __cosf(ang) : __sinf(ang);
  size_t o = ((size_t)b * SEQ_LEN + l) * D_MODEL + d;
  outF[o] = acc;
  outH[o] = (_Float16)acc;
}

// fused residual-add + LayerNorm over rows of [M,512]; emits f32 and/or f16
__global__ __launch_bounds__(256) void k_ln(const float* __restrict__ a, const float* __restrict__ res,
                                            const float* __restrict__ g, const float* __restrict__ beta,
                                            float* __restrict__ outF, _Float16* __restrict__ outH) {
  __shared__ float sred[16];
  int row = blockIdx.x, t = threadIdx.x;
  size_t base = (size_t)row * D_MODEL;
  float v0 = a[base + t], v1 = a[base + t + 256];
  if (res) { v0 += res[base + t]; v1 += res[base + t + 256]; }
  float s = v0 + v1, ss = v0 * v0 + v1 * v1;
#pragma unroll
  for (int m = 16; m >= 1; m >>= 1) { s += __shfl_xor(s, m); ss += __shfl_xor(ss, m); }
  int wv = t >> 5, lane = t & 31;
  if (lane == 0) { sred[wv] = s; sred[8 + wv] = ss; }
  __syncthreads();
  float S = 0.f, SS = 0.f;
#pragma unroll
  for (int w = 0; w < 8; ++w) { S += sred[w]; SS += sred[8 + w]; }
  float muv = S * (1.f / D_MODEL);
  float inv = rsqrtf(SS * (1.f / D_MODEL) - muv * muv + 1e-5f);
  float y0 = (v0 - muv) * inv * g[t] + beta[t];
  float y1 = (v1 - muv) * inv * g[t + 256] + beta[t + 256];
  if (outF) { outF[base + t] = y0; outF[base + t + 256] = y1; }
  if (outH) { outH[base + t] = (_Float16)y0; outH[base + t + 256] = (_Float16)y1; }
}

// stage one 32x32 B panel into LDS pre-swizzled in WMMA fragment order:
// element (k, n) -> sBuf[(n/16)*512 + ownerLane*16 + k%16], ownerLane = (k/16)*16 + n%16
__device__ __forceinline__ void stage_panel(_Float16* sBuf, const _Float16* wsrc, int N, int tid) {
#pragma unroll
  for (int p = tid; p < 512; p += 256) {
    int kk = p >> 4;             // 0..31
    int n2 = (p & 15) * 2;       // 0,2,..,30
    v2h w = *(const v2h*)(wsrc + kk * N + n2);
    int tile = n2 >> 4;
    int olane = ((kk >> 4) << 4) + (n2 & 15);
    int jj = kk & 15;
    sBuf[tile * 512 + olane * 16 + jj] = w.x;
    sBuf[tile * 512 + (olane + 1) * 16 + jj] = w.y;
  }
}

// ---------------------------------------------------------------- WMMA GEMM
// C[M,N] = act(A[M,K] @ Wm[K,N] + bias). A,Wm f16 row-major, N multiple of 32,
// K multiple of 32, M multiple of 128. 8 waves/block; block covers 128x32 of C;
// each wave owns a 16-row strip and two 16x16 tiles (A fragment reused).
// Double-buffered swizzled LDS B panels: one barrier per K-step, staging of
// panel i+1 overlaps the WMMAs of panel i. Optional transposed f16 output
// (outT) emits V^T in [B, H, HEAD_DIM, SEQ] layout for the attention kernel.
__global__ __launch_bounds__(256) void k_gemm32(const _Float16* __restrict__ A,
                                                const _Float16* __restrict__ Wm,
                                                const float* __restrict__ bias,
                                                float* __restrict__ outF, _Float16* __restrict__ outH,
                                                _Float16* __restrict__ outT,
                                                int N, int Nout, int K, int act) {
  __shared__ _Float16 sB[2 * 1024];  // 2 buffers x (two swizzled fragment tiles)
  int tid = threadIdx.x;
  int wv = tid >> 5, lane = tid & 31, hf = lane >> 4, lr = lane & 15;
  int n0 = blockIdx.x * 32;
  int m0 = blockIdx.y * 128 + wv * 16;
  v8f acc0 = {}, acc1 = {};
  const _Float16* wpn = Wm + n0;                       // next panel to stage
  const _Float16* arow = A + (size_t)(m0 + lr) * K;    // advances 32 cols / iter
  stage_panel(sB, wpn, N, tid);
  wpn += 32 * N;
  __syncthreads();
  int cur = 0;
  for (int k0 = 0; k0 < K; k0 += 32) {
    if (k0 + 32 < K) {  // stage next panel into the other buffer
      stage_panel(sB + (cur ^ 1) * 1024, wpn, N, tid);
      wpn += 32 * N;
      if (tid < 32 && k0 + 64 < K)
        __builtin_prefetch((const void*)(wpn + tid * N), 0, 1);
    }
    // ---- A fragment: two contiguous 8-half runs per lane
    v8h alo = *(const v8h*)(arow + hf * 16);
    v8h ahi = *(const v8h*)(arow + hf * 16 + 8);
    v16h a = __builtin_shufflevector(alo, ahi, 0, 1, 2, 3, 4, 5, 6, 7,
                                     8, 9, 10, 11, 12, 13, 14, 15);
    v16h b0 = *(const v16h*)(&sB[cur * 1024 + lane * 16]);
    v16h b1 = *(const v16h*)(&sB[cur * 1024 + 512 + lane * 16]);
    acc0 = __builtin_amdgcn_wmma_f32_16x16x32_f16(false, a, false, b0, (short)0, acc0, false, false);
    acc1 = __builtin_amdgcn_wmma_f32_16x16x32_f16(false, a, false, b1, (short)0, acc1, false, false);
    arow += 32;
    __syncthreads();
    cur ^= 1;
  }
  // ---- epilogue
  if (outT) {  // transposed f16 output: Vt[((b*H + h)*64 + dh)*512 + s]
#pragma unroll
    for (int tile = 0; tile < 2; ++tile) {
      const v8f& acc = tile ? acc1 : acc0;
      int n = n0 + tile * 16 + lr;
      int hh = n >> 6, dh = n & 63;
      int bb = m0 >> 9, sbase = (m0 & 511) + hf * 8;
      v8h hv;
#pragma unroll
      for (int r = 0; r < 8; ++r) {
        float v = acc[r] + (bias ? bias[n] : 0.f);
        hv[r] = (_Float16)v;
      }
      *(v8h*)(outT + (((size_t)bb * N_HEADS + hh) * HEAD_DIM + dh) * SEQ_LEN + sbase) = hv;
    }
    return;
  }
#pragma unroll
  for (int tile = 0; tile < 2; ++tile) {
    const v8f& acc = tile ? acc1 : acc0;
    int n = n0 + tile * 16 + lr;
    if (n >= Nout) continue;
    float bs = bias ? bias[n] : 0.f;
#pragma unroll
    for (int r = 0; r < 8; ++r) {
      int m = m0 + hf * 8 + r;
      float v = acc[r] + bs;
      if (act == 1) v = fmaxf(v, 0.f);
      else if (act == 2) {  // tanh-approx GELU
        float u = 0.7978845608f * (v + 0.044715f * v * v * v);
        v = 0.5f * v * (1.f + tanhf(u));
      }
      size_t o = (size_t)m * Nout + n;
      if (outF) outF[o] = v;
      if (outH) outH[o] = (_Float16)v;
    }
  }
}

// ------------------------------------------------------- fused flash attention
// One wave per (b, h, 16-row l-tile). Streams 32-wide S chunks:
// S = (QK^T)*tau + delta, causal mask, 1/sqrt(64), online softmax, O += P@V.
// Q/K fragments are contiguous b128 loads; V comes pre-transposed [B,H,64,S].
// P relayout goes through a double-buffered LDS tile: one barrier per chunk.
__global__ __launch_bounds__(32) void k_attn(const _Float16* __restrict__ Q,
                                             const _Float16* __restrict__ Km,
                                             const _Float16* __restrict__ Vt,
                                             const float* __restrict__ tau,
                                             const float* __restrict__ delta,
                                             _Float16* __restrict__ O, int causal) {
  __shared__ _Float16 sP[2 * 512];
  int lane = threadIdx.x & 31, hf = lane >> 4, lr = lane & 15;
  int m0 = blockIdx.x * 16, h = blockIdx.y, b = blockIdx.z;
  const size_t base = (size_t)b * SEQ_LEN * D_MODEL + h * HEAD_DIM;

  v16h aq0, aq1;
  {
    const _Float16* qr = Q + base + (size_t)(m0 + lr) * D_MODEL;
    v8h q0 = *(const v8h*)(qr + hf * 16);
    v8h q1 = *(const v8h*)(qr + hf * 16 + 8);
    v8h q2 = *(const v8h*)(qr + 32 + hf * 16);
    v8h q3 = *(const v8h*)(qr + 40 + hf * 16);
    aq0 = __builtin_shufflevector(q0, q1, 0, 1, 2, 3, 4, 5, 6, 7, 8, 9, 10, 11, 12, 13, 14, 15);
    aq1 = __builtin_shufflevector(q2, q3, 0, 1, 2, 3, 4, 5, 6, 7, 8, 9, 10, 11, 12, 13, 14, 15);
  }
  const _Float16* kp = Km + base + (size_t)lr * D_MODEL;      // row s0+lr
  const _Float16* vt0 = Vt + (((size_t)b * N_HEADS + h) * HEAD_DIM + lr) * SEQ_LEN;
  float tb = tau[b];
  v8f oacc[4] = {{}, {}, {}, {}};
  float rmax[8], rsum[8];
#pragma unroll
  for (int r = 0; r < 8; ++r) { rmax[r] = -INFINITY; rsum[r] = 0.f; }

  int bufp = 0;
  int sEnd = causal ? (m0 + 16) : SEQ_LEN;
  for (int s0 = 0; s0 < sEnd; s0 += 32) {
    v8f sc0 = {}, sc1 = {};
    const _Float16* kr0 = kp + (size_t)s0 * D_MODEL + hf * 16;
#pragma unroll
    for (int c = 0; c < 2; ++c) {  // head-dim chunks 0..31, 32..63
      v16h aq = c ? aq1 : aq0;
      v16h bk0 = *(const v16h*)(kr0 + c * 32);
      v16h bk1 = *(const v16h*)(kr0 + (size_t)16 * D_MODEL + c * 32);
      sc0 = __builtin_amdgcn_wmma_f32_16x16x32_f16(false, aq, false, bk0, (short)0, sc0, false, false);
      sc1 = __builtin_amdgcn_wmma_f32_16x16x32_f16(false, aq, false, bk1, (short)0, sc1, false, false);
    }
    float d0 = 0.f, d1 = 0.f;
    if (delta) {
      d0 = delta[(size_t)b * SEQ_LEN + s0 + lr];
      d1 = delta[(size_t)b * SEQ_LEN + s0 + 16 + lr];
    }
    float p0[8], p1[8];
#pragma unroll
    for (int r = 0; r < 8; ++r) {
      float x0 = (sc0[r] * tb + d0) * 0.125f;
      float x1 = (sc1[r] * tb + d1) * 0.125f;
      if (causal) {
        int mg = m0 + hf * 8 + r;
        if (s0 + lr > mg)      x0 = -INFINITY;
        if (s0 + 16 + lr > mg) x1 = -INFINITY;
      }
      float tmax = fmaxf(x0, x1);
#pragma unroll
      for (int msk = 8; msk >= 1; msk >>= 1) tmax = fmaxf(tmax, __shfl_xor(tmax, msk));
      float newm = fmaxf(rmax[r], tmax);
      float scale = (rmax[r] == -INFINITY) ? 0.f : __expf(rmax[r] - newm);
      float e0 = __expf(x0 - newm), e1 = __expf(x1 - newm);
      float ssum = e0 + e1;
#pragma unroll
      for (int msk = 8; msk >= 1; msk >>= 1) ssum += __shfl_xor(ssum, msk);
      rsum[r] = rsum[r] * scale + ssum;
      rmax[r] = newm;
#pragma unroll
      for (int t = 0; t < 4; ++t) oacc[t][r] *= scale;
      p0[r] = e0; p1[r] = e1;
    }
    _Float16* sp = sP + bufp * 512;
#pragma unroll
    for (int r = 0; r < 8; ++r) {
      sp[(hf * 8 + r) * 32 + lr]      = (_Float16)p0[r];
      sp[(hf * 8 + r) * 32 + 16 + lr] = (_Float16)p1[r];
    }
    __syncthreads();
    v8h plo = *(const v8h*)(&sp[lr * 32 + hf * 16]);
    v8h phi = *(const v8h*)(&sp[lr * 32 + hf * 16 + 8]);
    v16h ap = __builtin_shufflevector(plo, phi, 0, 1, 2, 3, 4, 5, 6, 7,
                                      8, 9, 10, 11, 12, 13, 14, 15);
#pragma unroll
    for (int t = 0; t < 4; ++t) {
      // B fragment of V chunk: dh = t*16 + lr row of V^T, contiguous in s
      v16h bv = *(const v16h*)(vt0 + (size_t)t * 16 * SEQ_LEN + s0 + hf * 16);
      oacc[t] = __builtin_amdgcn_wmma_f32_16x16x32_f16(false, ap, false, bv, (short)0, oacc[t], false, false);
    }
    bufp ^= 1;
  }
#pragma unroll
  for (int t = 0; t < 4; ++t)
#pragma unroll
    for (int r = 0; r < 8; ++r) {
      float val = oacc[t][r] / rsum[r];
      O[((size_t)b * SEQ_LEN + m0 + hf * 8 + r) * D_MODEL + h * HEAD_DIM + t * 16 + lr] = (_Float16)val;
    }
}

// de-normalize + slice last PRED_LEN
__global__ void k_out(const float* __restrict__ proj, const float* __restrict__ mean,
                      const float* __restrict__ stdv, float* __restrict__ out) {
  int idx = blockIdx.x * blockDim.x + threadIdx.x;
  if (idx >= B_SZ * PRED_LEN * C_OUT) return;
  int i = idx % C_OUT;
  int t = (idx / C_OUT) % PRED_LEN;
  int b = idx / (C_OUT * PRED_LEN);
  float v = proj[((size_t)b * DEC_LEN + LABEL_LEN + t) * C_OUT + i];
  out[idx] = v * stdv[b * ENC_IN + i] + mean[b * ENC_IN + i];
}

// ---------------------------------------------------------------- host driver
extern "C" void kernel_launch(void* const* d_in, const int* in_sizes, int n_in,
                              void* d_out, int out_size, void* d_ws, size_t ws_size,
                              hipStream_t stream) {
  (void)in_sizes; (void)n_in; (void)out_size; (void)ws_size;
  const float* x_enc      = (const float*)d_in[0];
  const float* x_mark_enc = (const float*)d_in[1];
  const float* x_mark_dec = (const float*)d_in[3];  // d_in[2] (x_dec) unused by reference

  const int TAU0 = 4, DELTA0 = 10, ENC_EMB = 16, DEC_EMB = 18, ENC_L0 = 20;
  const int DEC_L = 52, ENC_LN = 78, DEC_LN = 80, PROJ_W = 82, PROJ_B = 83;

  char* ws = (char*)d_ws;
  size_t off = 0;
  auto alloc = [&](size_t bytes) -> void* {
    off = (off + 255) & ~(size_t)255;
    void* p = ws + off;
    off += bytes;
    return p;
  };
  auto f32p = [&](int i) -> const float* { return (const float*)d_in[i]; };
  auto toH = [&](int i, int n) -> const _Float16* {
    _Float16* p = (_Float16*)alloc((size_t)n * sizeof(_Float16));
    k_f32_to_f16<<<dim3((n + 255) / 256), dim3(256), 0, stream>>>(f32p(i), p, n);
    return p;
  };

  const int M = B_SZ * SEQ_LEN;  // 8192 rows for both streams

  struct AttnW { const _Float16 *wq, *wk, *wv, *wo; int bq, bk, bv, bo; };
  auto attnW = [&](int base) -> AttnW {
    AttnW w;
    w.wq = toH(base + 0, D_MODEL * D_MODEL); w.bq = base + 1;
    w.wk = toH(base + 2, D_MODEL * D_MODEL); w.bk = base + 3;
    w.wv = toH(base + 4, D_MODEL * D_MODEL); w.bv = base + 5;
    w.wo = toH(base + 6, D_MODEL * D_MODEL); w.bo = base + 7;
    return w;
  };
  AttnW encA[2] = { attnW(ENC_L0), attnW(ENC_L0 + 16) };
  AttnW sa = attnW(DEC_L), ca = attnW(DEC_L + 8);

  const _Float16* fw1h[3]; const _Float16* fw2h[3];
  int fb1i[3], fb2i[3];
  int fbase[3] = { ENC_L0 + 8, ENC_L0 + 16 + 8, DEC_L + 16 };
  for (int i = 0; i < 3; ++i) {
    fw1h[i] = toH(fbase[i] + 0, D_MODEL * D_FF); fb1i[i] = fbase[i] + 1;
    fw2h[i] = toH(fbase[i] + 2, D_FF * D_MODEL); fb2i[i] = fbase[i] + 3;
  }
  _Float16* projPad = (_Float16*)alloc((size_t)D_MODEL * 32 * sizeof(_Float16));
  k_padw<<<dim3((D_MODEL * 32 + 255) / 256), dim3(256), 0, stream>>>(
      f32p(PROJ_W), projPad, D_MODEL, C_OUT, 32);

  float* meanB = (float*)alloc(B_SZ * ENC_IN * 4);
  float* stdB  = (float*)alloc(B_SZ * ENC_IN * 4);
  float* sTau  = (float*)alloc(B_SZ * ENC_IN * 4);
  float* sDel  = (float*)alloc(B_SZ * ENC_IN * 4);
  float* tauB  = (float*)alloc(B_SZ * 4);
  float* delB  = (float*)alloc((size_t)B_SZ * SEQ_LEN * 4);
  size_t actF = (size_t)M * D_MODEL * sizeof(float);
  size_t actH = (size_t)M * D_MODEL * sizeof(_Float16);
  float* X0 = (float*)alloc(actF); _Float16* X0h = (_Float16*)alloc(actH);
  float* X1 = (float*)alloc(actF); _Float16* X1h = (_Float16*)alloc(actH);
  float* X2 = (float*)alloc(actF); _Float16* X2h = (_Float16*)alloc(actH);
  float* Y  = (float*)alloc(actF);
  _Float16* Eh = (_Float16*)alloc(actH);
  _Float16* Qh = (_Float16*)alloc(actH);
  _Float16* Kh = (_Float16*)alloc(actH);
  _Float16* Vth = (_Float16*)alloc(actH);  // V^T  [B, H, 64, SEQ]
  _Float16* Ah = (_Float16*)alloc(actH);
  _Float16* Hh = (_Float16*)alloc((size_t)M * D_FF * sizeof(_Float16));
  float* Pf = (float*)alloc((size_t)M * C_OUT * sizeof(float));

  auto gemm = [&](const _Float16* A, const _Float16* Wm, const float* bias,
                  float* oF, _Float16* oH, _Float16* oT, int N, int Nout, int K, int act) {
    dim3 g(N / 32, M / 128);
    k_gemm32<<<g, dim3(256), 0, stream>>>(A, Wm, bias, oF, oH, oT, N, Nout, K, act);
  };
  auto attn = [&](const _Float16* q, const _Float16* k, const _Float16* vt,
                  const float* del, int causal, _Float16* o) {
    dim3 g(SEQ_LEN / 16, N_HEADS, B_SZ);
    k_attn<<<g, dim3(32), 0, stream>>>(q, k, vt, tauB, del, o, causal);
  };
  auto lnorm = [&](const float* a, const float* res, int gi, int bi, float* oF, _Float16* oH) {
    k_ln<<<dim3(M), dim3(256), 0, stream>>>(a, res, f32p(gi), f32p(bi), oF, oH);
  };

  // ---- de-stationarization stats + tau/delta projectors
  k_stats<<<dim3(B_SZ * ENC_IN), dim3(32), 0, stream>>>(x_enc, meanB, stdB);
  k_projconv<<<dim3(B_SZ * ENC_IN), dim3(32), 0, stream>>>(x_enc, f32p(TAU0 + 0), sTau);
  k_projconv<<<dim3(B_SZ * ENC_IN), dim3(32), 0, stream>>>(x_enc, f32p(DELTA0 + 0), sDel);
  k_projmlp<<<dim3(B_SZ), dim3(128), 0, stream>>>(sTau, stdB, f32p(TAU0 + 1), f32p(TAU0 + 2),
                                                  f32p(TAU0 + 3), f32p(TAU0 + 4), f32p(TAU0 + 5),
                                                  tauB, 1, 1);
  k_projmlp<<<dim3(B_SZ), dim3(128), 0, stream>>>(sDel, meanB, f32p(DELTA0 + 1), f32p(DELTA0 + 2),
                                                  f32p(DELTA0 + 3), f32p(DELTA0 + 4), f32p(DELTA0 + 5),
                                                  delB, SEQ_LEN, 0);

  // ---- encoder
  k_embed<<<dim3((SEQ_LEN * D_MODEL + 255) / 256, B_SZ), dim3(256), 0, stream>>>(
      x_enc, meanB, stdB, f32p(ENC_EMB), f32p(ENC_EMB + 1), x_mark_enc, X0, X0h, 0);
  for (int l = 0; l < 2; ++l) {
    const AttnW& w = encA[l];
    int base = ENC_L0 + 16 * l;
    gemm(X0h, w.wq, f32p(w.bq), nullptr, Qh, nullptr, D_MODEL, D_MODEL, D_MODEL, 0);
    gemm(X0h, w.wk, f32p(w.bk), nullptr, Kh, nullptr, D_MODEL, D_MODEL, D_MODEL, 0);
    gemm(X0h, w.wv, f32p(w.bv), nullptr, nullptr, Vth, D_MODEL, D_MODEL, D_MODEL, 0);
    attn(Qh, Kh, Vth, delB, 0, Ah);
    gemm(Ah, w.wo, f32p(w.bo), Y, nullptr, nullptr, D_MODEL, D_MODEL, D_MODEL, 0);
    lnorm(X0, Y, base + 12, base + 13, X1, X1h);
    gemm(X1h, fw1h[l], f32p(fb1i[l]), nullptr, Hh, nullptr, D_FF, D_FF, D_MODEL, 2);
    gemm(Hh, fw2h[l], f32p(fb2i[l]), Y, nullptr, nullptr, D_MODEL, D_MODEL, D_FF, 0);
    lnorm(X1, Y, base + 14, base + 15, X0, X0h);
  }
  lnorm(X0, nullptr, ENC_LN, ENC_LN + 1, nullptr, Eh);  // encoder memory (f16 only)

  // ---- decoder
  k_embed<<<dim3((SEQ_LEN * D_MODEL + 255) / 256, B_SZ), dim3(256), 0, stream>>>(
      x_enc, meanB, stdB, f32p(DEC_EMB), f32p(DEC_EMB + 1), x_mark_dec, X0, X0h, 1);
  // self-attention (causal, delta = 0)
  gemm(X0h, sa.wq, f32p(sa.bq), nullptr, Qh, nullptr, D_MODEL, D_MODEL, D_MODEL, 0);
  gemm(X0h, sa.wk, f32p(sa.bk), nullptr, Kh, nullptr, D_MODEL, D_MODEL, D_MODEL, 0);
  gemm(X0h, sa.wv, f32p(sa.bv), nullptr, nullptr, Vth, D_MODEL, D_MODEL, D_MODEL, 0);
  attn(Qh, Kh, Vth, nullptr, 1, Ah);
  gemm(Ah, sa.wo, f32p(sa.bo), Y, nullptr, nullptr, D_MODEL, D_MODEL, D_MODEL, 0);
  lnorm(X0, Y, DEC_L + 20, DEC_L + 21, X1, X1h);
  // cross-attention (K/V from encoder memory)
  gemm(X1h, ca.wq, f32p(ca.bq), nullptr, Qh, nullptr, D_MODEL, D_MODEL, D_MODEL, 0);
  gemm(Eh,  ca.wk, f32p(ca.bk), nullptr, Kh, nullptr, D_MODEL, D_MODEL, D_MODEL, 0);
  gemm(Eh,  ca.wv, f32p(ca.bv), nullptr, nullptr, Vth, D_MODEL, D_MODEL, D_MODEL, 0);
  attn(Qh, Kh, Vth, delB, 0, Ah);
  gemm(Ah, ca.wo, f32p(ca.bo), Y, nullptr, nullptr, D_MODEL, D_MODEL, D_MODEL, 0);
  lnorm(X1, Y, DEC_L + 22, DEC_L + 23, X2, X2h);
  // FFN
  gemm(X2h, fw1h[2], f32p(fb1i[2]), nullptr, Hh, nullptr, D_FF, D_FF, D_MODEL, 2);
  gemm(Hh, fw2h[2], f32p(fb2i[2]), Y, nullptr, nullptr, D_MODEL, D_MODEL, D_FF, 0);
  lnorm(X2, Y, DEC_L + 24, DEC_L + 25, X0, X0h);
  // final LN + output projection (padded to N=32) + de-normalization
  lnorm(X0, nullptr, DEC_LN, DEC_LN + 1, nullptr, X1h);
  gemm(X1h, projPad, f32p(PROJ_B), Pf, nullptr, nullptr, 32, C_OUT, D_MODEL, 0);
  k_out<<<dim3((B_SZ * PRED_LEN * C_OUT + 255) / 256), dim3(256), 0, stream>>>(
      Pf, meanB, stdB, (float*)d_out);
}
